// MixtureOfExperts_64699387347184
// MI455X (gfx1250) — matrix operand order repaired and checked by reading, not compiled, not run on hardware
//
#include <hip/hip_runtime.h>
#include <math.h>

// Problem constants (from reference setup): B=4, S=4096 -> T=16384, D=1024, F=2048, E=8, top-2.
#define T_TOK 16384
#define DIM   1024
#define FFN   2048
#define NEXP  8

typedef __attribute__((ext_vector_type(16))) __bf16 v16bf;
typedef __attribute__((ext_vector_type(8)))  __bf16 v8bf;
typedef __attribute__((ext_vector_type(8)))  float  v8f;

union FragAB { v16bf v; v8bf h[2]; };

// A-matrix 16x32 bf16 fragment (ISA 7.12.2): lane holds row m = lane&15;
// elements 0..7 -> K = k0 + 8*hi + i, elements 8..15 -> K = k0 + 8*hi + 16 + (i-8).
__device__ __forceinline__ v16bf load_a_frag(const __bf16* p) {
  FragAB f;
  f.h[0] = *reinterpret_cast<const v8bf*>(p);
  f.h[1] = *reinterpret_cast<const v8bf*>(p + 16);
  return f.v;
}
// B-matrix 32x16 bf16 fragment: lane holds column n = lane&15;
// elements 0..7 -> K = k0 + 8*hi + i, elements 8..15 -> K = k0 + 8*hi + 16 + (i-8).
__device__ __forceinline__ v16bf load_b_frag(const __bf16* p) {
  FragAB f;
  f.h[0] = *reinterpret_cast<const v8bf*>(p);
  f.h[1] = *reinterpret_cast<const v8bf*>(p + 16);
  return f.v;
}

__device__ __forceinline__ float gelu_exact(float x) {
  return 0.5f * x * (1.0f + erff(x * 0.70710678118654752440f));
}

// ---------------- f32 -> bf16 conversion ----------------
__global__ void moe_cvt_bf16(const float* __restrict__ src, __bf16* __restrict__ dst, int n) {
  int i = blockIdx.x * blockDim.x + threadIdx.x;
  int stride = gridDim.x * blockDim.x;
  for (; i < n; i += stride) dst[i] = (__bf16)src[i];
}

// ---------------- router: logits, softmax, top-2, expert lists, usage partials ----------------
// grid = 256 blocks x 256 threads; each wave32 handles 8 tokens (64 tokens/block).
__global__ __launch_bounds__(256) void moe_router(
    const float* __restrict__ x, const float* __restrict__ wr,
    int* __restrict__ cnt, int* __restrict__ idxl, float* __restrict__ gatel,
    float* __restrict__ usage_part) {
  __shared__ float s_wr[NEXP * DIM];   // 32 KB
  __shared__ float s_u[8][NEXP];
  for (int i = threadIdx.x; i < NEXP * DIM; i += 256) s_wr[i] = wr[i];
  __syncthreads();

  const int wave = threadIdx.x >> 5, lane = threadIdx.x & 31;
  float uacc[NEXP];
  #pragma unroll
  for (int e = 0; e < NEXP; ++e) uacc[e] = 0.f;

  const int tok0 = blockIdx.x * 64 + wave * 8;
  for (int i = 0; i < 8; ++i) {
    const int t = tok0 + i;
    const float* xr = x + (size_t)t * DIM;
    float acc[NEXP];
    #pragma unroll
    for (int e = 0; e < NEXP; ++e) acc[e] = 0.f;
    for (int d = lane; d < DIM; d += 32) {
      float xv = xr[d];
      #pragma unroll
      for (int e = 0; e < NEXP; ++e) acc[e] += xv * s_wr[e * DIM + d];
    }
    #pragma unroll
    for (int e = 0; e < NEXP; ++e) {
      float v = acc[e];
      for (int off = 16; off > 0; off >>= 1) v += __shfl_xor(v, off, 32);
      acc[e] = v;
    }
    // softmax over 8 logits (all lanes redundantly, f32)
    float mx = acc[0];
    #pragma unroll
    for (int e = 1; e < NEXP; ++e) mx = fmaxf(mx, acc[e]);
    float p[NEXP], s = 0.f;
    #pragma unroll
    for (int e = 0; e < NEXP; ++e) { p[e] = expf(acc[e] - mx); s += p[e]; }
    float inv = 1.f / s;
    #pragma unroll
    for (int e = 0; e < NEXP; ++e) p[e] *= inv;
    // top-2, earliest index wins ties (matches lax.top_k)
    int i1 = 0;
    #pragma unroll
    for (int e = 1; e < NEXP; ++e) if (p[e] > p[i1]) i1 = e;
    int i2 = (i1 == 0) ? 1 : 0;
    #pragma unroll
    for (int e = 0; e < NEXP; ++e) if (e != i1 && p[e] > p[i2]) i2 = e;
    float rn = 1.f / (p[i1] + p[i2]);

    if (lane == 0) {
      #pragma unroll
      for (int e = 0; e < NEXP; ++e) uacc[e] += p[e];   // fixed token order
      int pos1 = atomicAdd(&cnt[i1], 1);
      idxl[i1 * T_TOK + pos1] = t;  gatel[i1 * T_TOK + pos1] = p[i1] * rn;
      int pos2 = atomicAdd(&cnt[i2], 1);
      idxl[i2 * T_TOK + pos2] = t;  gatel[i2 * T_TOK + pos2] = p[i2] * rn;
    }
  }
  if (lane == 0) {
    #pragma unroll
    for (int e = 0; e < NEXP; ++e) s_u[wave][e] = uacc[e];
  }
  __syncthreads();
  if (threadIdx.x < NEXP) {
    float s2 = 0.f;
    for (int w = 0; w < 8; ++w) s2 += s_u[w][threadIdx.x];  // fixed order
    usage_part[blockIdx.x * NEXP + threadIdx.x] = s2;
  }
}

// ---------------- aux loss: deterministic final reduction ----------------
__global__ void moe_aux(const float* __restrict__ usage_part, float* __restrict__ out_aux) {
  __shared__ float s_m[NEXP];
  if (threadIdx.x < NEXP) {
    float s = 0.f;
    for (int b = 0; b < 256; ++b) s += usage_part[b * NEXP + threadIdx.x]; // fixed order
    s_m[threadIdx.x] = s * (1.0f / (float)T_TOK);
  }
  __syncthreads();
  if (threadIdx.x == 0) {
    float a = 0.f;
    #pragma unroll
    for (int e = 0; e < NEXP; ++e) a += s_m[e] * s_m[e];
    out_aux[0] = (float)NEXP * a;
  }
}

// ---------------- fused expert FFN: gather -> GEMM1 -> GELU -> GEMM2 -> gated scatter ----------------
// grid = NEXP * 1024 tiles of 16 tokens (worst case), early-exit beyond per-expert count.
// block = 256 threads (8 wave32). LDS: x tile 32KB + h tile 64KB + token/gate lists.
// N-dimension blocked 4-wide so one A-fragment feeds 4 WMMAs (2.5 mem-ops/wmma).
__global__ __launch_bounds__(256) void moe_ffn(
    const __bf16* __restrict__ xbf, const __bf16* __restrict__ w1bf,
    const __bf16* __restrict__ w2bf, const int* __restrict__ cnt,
    const int* __restrict__ idxl, const float* __restrict__ gatel,
    float* __restrict__ out) {
  const int e = blockIdx.x >> 10;
  const int tile = blockIdx.x & 1023;
  const int start = tile * 16;
  const int n_e = cnt[e];
  if (start >= n_e) return;                 // uniform across block
  const int rem = min(16, n_e - start);

  extern __shared__ char smem[];
  __bf16* s_x   = (__bf16*)smem;                        // [16][1024] bf16 = 32768 B
  __bf16* s_h   = (__bf16*)(smem + 32768);              // [16][2048] bf16 = 65536 B
  int*    s_tok = (int*)(smem + 32768 + 65536);         // 64 B
  float*  s_gate= (float*)(smem + 32768 + 65536 + 64);  // 64 B

  if (threadIdx.x < 16) {
    int m = threadIdx.x;
    int src = (m < rem) ? (start + m) : start;          // pad with row 0's token, gate 0
    s_tok[m]  = idxl[e * T_TOK + src];
    s_gate[m] = (m < rem) ? gatel[e * T_TOK + src] : 0.f;
  }
  __syncthreads();

  // Stage 16 gathered bf16 x rows into LDS (16B vector copies).
  {
    const int row = threadIdx.x >> 4;
    const int sub = threadIdx.x & 15;                   // 64 bf16 (128 B) per thread
    const uint4* s4 = reinterpret_cast<const uint4*>(xbf + (size_t)s_tok[row] * DIM + sub * 64);
    uint4* d4 = reinterpret_cast<uint4*>(s_x + row * DIM + sub * 64);
    #pragma unroll
    for (int j = 0; j < 8; ++j) d4[j] = s4[j];
  }
  __syncthreads();

  const int wave = threadIdx.x >> 5;
  const int lane = threadIdx.x & 31;
  const int lnr  = lane & 15;
  const int hi   = lane >> 4;
  const __bf16* w1e = w1bf + (size_t)e * FFN * DIM;     // [F][D] row-major == B^T
  const __bf16* w2e = w2bf + (size_t)e * DIM * FFN;     // [D][F] row-major == B^T

  // GEMM1: h[16, 2048] = x_tile @ w1^T; each wave: 16 N-tiles as 4 groups of 4.
  for (int g = 0; g < 4; ++g) {
    const int nc0 = wave * 256 + g * 64;                // first f column of this 4-tile group
    v8f acc[4] = {};
    const __bf16* arow = s_x + lnr * DIM + hi * 8;
    const __bf16* brow[4];
    #pragma unroll
    for (int j = 0; j < 4; ++j)
      brow[j] = w1e + (size_t)(nc0 + j * 16 + lnr) * DIM + hi * 8;
    for (int k0 = 0; k0 < DIM; k0 += 32) {
      v16bf a = load_a_frag(arow + k0);                 // one A frag -> 4 WMMAs
      #pragma unroll
      for (int j = 0; j < 4; ++j)
        acc[j] = __builtin_amdgcn_wmma_f32_16x16x32_bf16(
            false, a, false, load_b_frag(brow[j] + k0), (short)0, acc[j], false, false);
    }
    #pragma unroll
    for (int j = 0; j < 4; ++j) {
      const int c = nc0 + j * 16 + lnr;                 // C layout: N = lane&15, M = 8*hi + r
      #pragma unroll
      for (int r = 0; r < 8; ++r) {
        int m = hi * 8 + r;
        s_h[m * FFN + c] = (__bf16)gelu_exact(acc[j][r]);
      }
    }
  }
  __syncthreads();

  // GEMM2: y[16, 1024] = h @ w2^T; each wave: 8 N-tiles as 2 groups of 4; gated atomic scatter.
  for (int g = 0; g < 2; ++g) {
    const int d0 = wave * 128 + g * 64;
    v8f acc[4] = {};
    const __bf16* arow = s_h + lnr * FFN + hi * 8;
    const __bf16* brow[4];
    #pragma unroll
    for (int j = 0; j < 4; ++j)
      brow[j] = w2e + (size_t)(d0 + j * 16 + lnr) * FFN + hi * 8;
    for (int k0 = 0; k0 < FFN; k0 += 32) {
      v16bf a = load_a_frag(arow + k0);
      #pragma unroll
      for (int j = 0; j < 4; ++j)
        acc[j] = __builtin_amdgcn_wmma_f32_16x16x32_bf16(
            false, a, false, load_b_frag(brow[j] + k0), (short)0, acc[j], false, false);
    }
    #pragma unroll
    for (int j = 0; j < 4; ++j) {
      #pragma unroll
      for (int r = 0; r < 8; ++r) {
        int m = hi * 8 + r;
        if (m < rem)
          atomicAdd(out + (size_t)s_tok[m] * DIM + d0 + j * 16 + lnr, acc[j][r] * s_gate[m]);
      }
    }
  }
}

// ---------------- launch ----------------
extern "C" void kernel_launch(void* const* d_in, const int* in_sizes, int n_in,
                              void* d_out, int out_size, void* d_ws, size_t ws_size,
                              hipStream_t stream) {
  const float* x  = (const float*)d_in[0];   // [T, D]
  const float* wr = (const float*)d_in[1];   // [E, D]
  const float* w1 = (const float*)d_in[2];   // [E, F, D]
  const float* w2 = (const float*)d_in[3];   // [E, D, F]
  float* out = (float*)d_out;                // [T*D] out ++ [1] aux

  // Workspace layout (needs ~101.8 MB):
  char* ws = (char*)d_ws;
  __bf16* xbf   = (__bf16*)(ws);                    // 32 MB
  __bf16* w1bf  = (__bf16*)(ws + 33554432);         // 32 MB
  __bf16* w2bf  = (__bf16*)(ws + 67108864);         // 32 MB
  int*    idxl  = (int*)   (ws + 100663296);        // 8*T ints
  float*  gatel = (float*) (ws + 101187584);        // 8*T floats
  int*    cnt   = (int*)   (ws + 101711872);        // 8 (padded to 256B)
  float*  upart = (float*) (ws + 101712128);        // 256 blocks * 8

  hipMemsetAsync(cnt, 0, 256, stream);
  hipMemsetAsync(d_out, 0, (size_t)T_TOK * DIM * sizeof(float), stream);

  moe_cvt_bf16<<<4096, 256, 0, stream>>>(x,  xbf,  T_TOK * DIM);
  moe_cvt_bf16<<<4096, 256, 0, stream>>>(w1, w1bf, NEXP * FFN * DIM);
  moe_cvt_bf16<<<4096, 256, 0, stream>>>(w2, w2bf, NEXP * DIM * FFN);

  moe_router<<<256, 256, 0, stream>>>(x, wr, cnt, idxl, gatel, upart);
  moe_aux<<<1, 32, 0, stream>>>(upart, out + (size_t)T_TOK * DIM);

  const int smem = 32768 + 65536 + 64 + 64;   // 98432 B dynamic LDS
  moe_ffn<<<NEXP * 1024, 256, smem, stream>>>(xbf, w1bf, w2bf, cnt, idxl, gatel, out);
}
